// SelfAttentionLayer_62294205661728
// MI455X (gfx1250) — compile-verified, matrix operand order checked by reading, not run on hardware
//
#include <hip/hip_runtime.h>
#include <hip/hip_bf16.h>

typedef __attribute__((ext_vector_type(16))) _Float16 v16h;
typedef __attribute__((ext_vector_type(8)))  _Float16 v8h;
typedef __attribute__((ext_vector_type(8)))  float    v8f;

typedef __attribute__((ext_vector_type(4))) unsigned tdm_v4u;
typedef __attribute__((ext_vector_type(8))) int      tdm_v8i;
typedef __attribute__((ext_vector_type(4))) int      tdm_v4i;

#if __has_builtin(__builtin_amdgcn_tensor_load_to_lds)
#define HAVE_TDM 1
#endif

// ---------------------------------------------------------------------------
// Async copy 16B global -> LDS (CDNA5 ASYNCcnt path, no VGPR round-trip).
// LDS address = low 32 bits of the generic pointer (LDS aperture truncation).
// ---------------------------------------------------------------------------
__device__ __forceinline__ void async_b128(const _Float16* gsrc, _Float16* ldst) {
  unsigned loff = (unsigned)(uintptr_t)(void*)ldst;
  asm volatile("global_load_async_to_lds_b128 %0, %1, off"
               :: "v"(loff), "v"(gsrc) : "memory");
}
__device__ __forceinline__ void wait_async0() {
  asm volatile("s_wait_asynccnt 0x0" ::: "memory");
}

// ---------------------------------------------------------------------------
// Fragment loader matching the CDNA5 16-bit A/B WMMA layout (wave32):
//   lanes 0-15 : row r = lane,    halves 0-7 = K 0-7,  halves 8-15 = K 16-23
//   lanes16-31 : row r = lane-16, halves 0-7 = K 8-15, halves 8-15 = K 24-31
// ---------------------------------------------------------------------------
__device__ __forceinline__ v16h load_frag_ld(const _Float16* p, int ld) {
  const int l = threadIdx.x & 31;
  const int r = l & 15;
  const int g = l >> 4;
  const _Float16* base = p + r * ld + g * 8;
  v8h lo = *(const v8h*)(base);
  v8h hi = *(const v8h*)(base + 16);
  return __builtin_shufflevector(lo, hi, 0,1,2,3,4,5,6,7,8,9,10,11,12,13,14,15);
}

__device__ __forceinline__ v8f wmma_f16(v16h a, v16h b, v8f c) {
  return __builtin_amdgcn_wmma_f32_16x16x32_f16(
      false, a, false, b, (short)0, c, false, false);
}

// ---------------------------------------------------------------------------
// fp32 -> f16 conversion (plain, and transposing variant for weights)
// ---------------------------------------------------------------------------
__global__ void cvt_f32_to_f16(const float* __restrict__ in,
                               _Float16* __restrict__ out, int n) {
  int i = blockIdx.x * blockDim.x + threadIdx.x;
  int stride = gridDim.x * blockDim.x;
  for (; i < n; i += stride) out[i] = (_Float16)in[i];
}

// in: [K,N] row-major f32  ->  out: [N,K] row-major f16
__global__ void cvt_f32_to_f16_T(const float* __restrict__ in,
                                 _Float16* __restrict__ out, int K, int N) {
  int i = blockIdx.x * blockDim.x + threadIdx.x;
  int total = K * N;
  int stride = gridDim.x * blockDim.x;
  for (; i < total; i += stride) {
    int k = i / N, n = i - k * N;
    out[(size_t)n * K + k] = (_Float16)in[i];
  }
}

// ---------------------------------------------------------------------------
// Tiled WMMA GEMM: C[M,N] = A[M,K](f16) * Bt[N,K](f16, pre-transposed) + bias
// BM=BN=128, BK=32, 256 threads = 8 waves (4M x 2N), wave tile 32x64.
// Both tiles are contiguous-K row-major copies -> staged entirely with
// global_load_async_to_lds_b128, double-buffered, s_wait_asynccnt sync.
// ---------------------------------------------------------------------------
#define BM 128
#define BN 128
#define BK 32
#define LDT 40  // BK + 8-half pad: 80B row stride, keeps 16B alignment

__global__ __launch_bounds__(256)
void gemm_f16_kernel(const _Float16* __restrict__ A, const _Float16* __restrict__ Bt,
                     const float* __restrict__ bias,
                     float* __restrict__ Cf, _Float16* __restrict__ Ch,
                     int M, int N, int K, int out_half) {
  __shared__ __align__(16) _Float16 sA[2][BM][LDT];
  __shared__ __align__(16) _Float16 sB[2][BN][LDT];  // [n][k]

  const int tid  = threadIdx.x;
  const int lane = tid & 31;
  const int wid  = tid >> 5;
  const int wm   = wid >> 1;   // 0..3
  const int wn   = wid & 1;    // 0..1
  const int n0   = blockIdx.x * BN;
  const int m0   = blockIdx.y * BM;

  v8f acc[2][4];
  for (int i = 0; i < 2; ++i)
    for (int j = 0; j < 4; ++j)
      for (int e = 0; e < 8; ++e) acc[i][j][e] = 0.f;

  const int nk = K / BK;

  auto stage = [&](int kt, int buf) {
    for (int it = 0; it < 2; ++it) {
      int c = tid + it * 256;          // 512 16B-chunks per 128x32 tile
      int row = c >> 2, kc = (c & 3) * 8;
      async_b128(A  + (size_t)(m0 + row) * K + kt * BK + kc, &sA[buf][row][kc]);
      async_b128(Bt + (size_t)(n0 + row) * K + kt * BK + kc, &sB[buf][row][kc]);
    }
  };

  stage(0, 0);
  wait_async0();
  __syncthreads();

  for (int kt = 0; kt < nk; ++kt) {
    const int cur = kt & 1;
    if (kt + 1 < nk) stage(kt + 1, cur ^ 1);  // DMA into other buffer

    v16h af[2], bf[4];
    for (int i = 0; i < 2; ++i)
      af[i] = load_frag_ld(&sA[cur][wm * 32 + i * 16][0], LDT);
    for (int j = 0; j < 4; ++j)
      bf[j] = load_frag_ld(&sB[cur][wn * 64 + j * 16][0], LDT);
    for (int i = 0; i < 2; ++i)
      for (int j = 0; j < 4; ++j)
        acc[i][j] = wmma_f16(af[i], bf[j], acc[i][j]);

    if (kt + 1 < nk) {
      wait_async0();
      __syncthreads();
    }
  }

  // epilogue: C layout -> lane (g=lane>>4, n=lane&15), VGPR e holds row 8g+e
  const int g = lane >> 4, nl = lane & 15;
  for (int i = 0; i < 2; ++i) {
    for (int j = 0; j < 4; ++j) {
      int colg = n0 + wn * 64 + j * 16 + nl;
      float bv = bias ? bias[colg] : 0.f;
      for (int e = 0; e < 8; ++e) {
        int rowg = m0 + wm * 32 + i * 16 + 8 * g + e;
        float v = acc[i][j][e] + bv;
        if (out_half) Ch[(size_t)rowg * N + colg] = (_Float16)v;
        else          Cf[(size_t)rowg * N + colg] = v;
      }
    }
  }
}

// ---------------------------------------------------------------------------
// Flash attention (causal). kqv: f16 [B,T,3C] rows = {k | q | v}.
// 1 workgroup = one (batch, 16-query-row tile); 4 waves, wave w owns
// V columns [w*256, w*256+256). 32-key chunks, online softmax.
// Q tile loaded by the Tensor Data Mover; K chunks by async-to-LDS.
// NOTE: reference scales scores by 1/sqrt(T), not 1/sqrt(C).
// ---------------------------------------------------------------------------
#define ATT_C  1024
#define ATT_3C 3072
#define VLD    40   // transposed-V pad: [col][key], 80B stride

__global__ __launch_bounds__(128)
void attn_kernel(const _Float16* __restrict__ kqv, _Float16* __restrict__ out, int T_) {
  extern __shared__ char smem[];
  _Float16* qs  = (_Float16*)(smem);            // [16][1024]   32 KB
  _Float16* ks  = (_Float16*)(smem + 32768);    // [32][1024]   64 KB
  _Float16* vT  = (_Float16*)(smem + 98304);    // [1024][40]   80 KB
  _Float16* Ps  = (_Float16*)(smem + 180224);   // [16][32]      1 KB
  float*    Ss  = (float*)   (smem + 181248);   // [16][32]      2 KB
  float*    mst = (float*)   (smem + 183296);   // [16] running max
  float*    lst = (float*)   (smem + 183360);   // [16] running sum
  float*    cst = (float*)   (smem + 183424);   // [16] correction

  const int tid  = threadIdx.x;  // 0..127
  const int wid  = tid >> 5;     // 0..3
  const int lane = tid & 31;
  const int T16  = T_ >> 4;
  const int b    = blockIdx.x / T16;
  const int m0   = (blockIdx.x % T16) * 16;
  const _Float16* base = kqv + (size_t)b * T_ * ATT_3C;

  // ---- stage Q tile [16][1024] (q = column offset C inside kqv rows) ----
#ifdef HAVE_TDM
  if (wid == 0) {
    unsigned lds_addr = (unsigned)(uintptr_t)(void*)qs;
    unsigned long long ga =
        (unsigned long long)(uintptr_t)(base + (size_t)m0 * ATT_3C + ATT_C);
    // D# group0: count=1 | lds_addr | global_addr[56:0] | type=2
    tdm_v4u g0 = { 1u, lds_addr, (unsigned)ga,
                   (unsigned)((ga >> 32) & 0x01FFFFFFu) | (2u << 30) };
    // D# group1: data_size=2B; tensor_dim0=1024; tensor_dim1=16;
    //            tile_dim0=1024; tile_dim1=16; tensor_dim0_stride=3072
    tdm_v8i g1 = { (int)(1u << 16),          // [17:16] data_size = 1 (2B)
                   (int)(1024u << 16),       // [79:48] tensor_dim0 (low 16 here)
                   (int)(16u << 16),         // [111:80] tensor_dim1
                   (int)(1024u << 16),       // [127:112] tile_dim0
                   16,                       // [143:128] tile_dim1
                   3072,                     // [207:160] tensor_dim0_stride
                   0, 0 };
    tdm_v4i gz = { 0, 0, 0, 0 };
#if __has_include(<hip/amd_detail/amd_gfx1250_TDM.h>)
    tdm_v8i gz8 = { 0, 0, 0, 0, 0, 0, 0, 0 };
    __builtin_amdgcn_tensor_load_to_lds(g0, g1, gz, gz, gz8, 0);
#else
    __builtin_amdgcn_tensor_load_to_lds(g0, g1, gz, gz, 0);
#endif
    __builtin_amdgcn_s_wait_tensorcnt(0);
  }
#else
  for (int idx = tid; idx < 16 * (ATT_C / 8); idx += 128) {
    int r = idx >> 7, c8 = (idx & 127) * 8;
    async_b128(base + (size_t)(m0 + r) * ATT_3C + ATT_C + c8, qs + r * ATT_C + c8);
  }
  wait_async0();
#endif
  if (tid < 16) { mst[tid] = -3.0e38f; lst[tid] = 0.f; }
  __syncthreads();

  v8f acc[16];
  for (int t = 0; t < 16; ++t)
    for (int e = 0; e < 8; ++e) acc[t][e] = 0.f;

  const int   col0     = wid * 256;
  const float sm_scale = rsqrtf((float)T_);
  const int   kmax     = m0 + 16;  // keys 0..m0+15 needed

  for (int kk = 0; kk < kmax; kk += 32) {
    // K chunk: async DMA straight to LDS. V chunk: regs + transposed stores.
    for (int idx = tid; idx < 32 * (ATT_C / 8); idx += 128) {
      int r = idx >> 7;                // key row within chunk
      int c8 = (idx & 127) * 8;
      const _Float16* rowp = base + (size_t)(kk + r) * ATT_3C;
      async_b128(rowp + c8, ks + r * ATT_C + c8);                  // k @ +0
      v8h vv = *(const v8h*)(rowp + 2 * ATT_C + c8);               // v @ +2C
      for (int j = 0; j < 8; ++j) vT[(c8 + j) * VLD + r] = vv[j];
    }
    if (kk + 32 < kmax)  // warm L2 for next chunk
      __builtin_prefetch(base + (size_t)(kk + 32 + (tid & 31)) * ATT_3C, 0, 1);
    wait_async0();
    __syncthreads();

    // S = Q * K^T over C=1024 (redundant per wave; identical results)
    v8f s0, s1;
    for (int e = 0; e < 8; ++e) { s0[e] = 0.f; s1[e] = 0.f; }
    for (int c = 0; c < ATT_C; c += 32) {
      v16h qa  = load_frag_ld(qs + c, ATT_C);
      v16h kb0 = load_frag_ld(ks + c, ATT_C);
      v16h kb1 = load_frag_ld(ks + 16 * ATT_C + c, ATT_C);
      s0 = wmma_f16(qa, kb0, s0);
      s1 = wmma_f16(qa, kb1, s1);
    }
    if (wid == 0) {
      int g = lane >> 4, n = lane & 15;
      for (int e = 0; e < 8; ++e) {
        Ss[(8 * g + e) * 32 + n]      = s0[e] * sm_scale;
        Ss[(8 * g + e) * 32 + 16 + n] = s1[e] * sm_scale;
      }
    }
    __syncthreads();

    // online softmax bookkeeping: one thread per query row
    if (tid < 16) {
      const int r = tid, mg = m0 + r;
      float om = mst[r], cm = om;
      float sv[32];
      for (int j = 0; j < 32; ++j) {
        int key = kk + j;
        float s = (key <= mg) ? Ss[r * 32 + j] : -3.0e38f;  // causal mask
        sv[j] = s;
        cm = fmaxf(cm, s);
      }
      float corr = __expf(om - cm);
      float l = lst[r] * corr;
      for (int j = 0; j < 32; ++j) {
        float p = __expf(sv[j] - cm);
        Ps[r * 32 + j] = (_Float16)p;
        l += p;
      }
      mst[r] = cm; lst[r] = l; cst[r] = corr;
    }
    __syncthreads();

    // rescale accumulators, then acc += P(16x32) * V(32 x cols)
    {
      int g = lane >> 4;
      float f[8];
      for (int e = 0; e < 8; ++e) f[e] = cst[8 * g + e];
      for (int t = 0; t < 16; ++t)
        for (int e = 0; e < 8; ++e) acc[t][e] *= f[e];
      v16h pa = load_frag_ld(Ps, 32);
      for (int t = 0; t < 16; ++t) {
        v16h vb = load_frag_ld(vT + (size_t)(col0 + t * 16) * VLD, VLD);
        acc[t] = wmma_f16(pa, vb, acc[t]);
      }
    }
    __syncthreads();  // protect ks/vT/Ps before next chunk overwrite
  }

  // finalize: divide by softmax sum, emit f16 for the projection GEMM
  {
    int g = lane >> 4, n = lane & 15;
    float linv[8];
    for (int e = 0; e < 8; ++e) linv[e] = 1.0f / lst[8 * g + e];
    for (int t = 0; t < 16; ++t) {
      int col = col0 + t * 16 + n;
      for (int e = 0; e < 8; ++e) {
        int row = m0 + 8 * g + e;
        out[(size_t)(b * T_ + row) * ATT_C + col] = (_Float16)(acc[t][e] * linv[e]);
      }
    }
  }
}

// ---------------------------------------------------------------------------
// Host launcher
// ---------------------------------------------------------------------------
extern "C" void kernel_launch(void* const* d_in, const int* in_sizes, int n_in,
                              void* d_out, int out_size, void* d_ws, size_t ws_size,
                              hipStream_t stream) {
  (void)in_sizes; (void)n_in; (void)out_size; (void)ws_size;
  const float* x      = (const float*)d_in[0];
  const float* w_kqv  = (const float*)d_in[1];
  const float* b_kqv  = (const float*)d_in[2];
  const float* w_proj = (const float*)d_in[3];
  const float* b_proj = (const float*)d_in[4];
  float* out = (float*)d_out;

  const int T = 2048, C = 1024, Bb = 4;
  const int Mrows = Bb * T;  // 8192

  char* ws = (char*)d_ws;  // 88 MB total
  _Float16* xh     = (_Float16*)(ws);                               // 16 MB
  _Float16* wkqvT  = (_Float16*)(ws + (size_t)16 * 1024 * 1024);    //  6 MB  [3C][C]
  _Float16* wprojT = (_Float16*)(ws + (size_t)22 * 1024 * 1024);    //  2 MB  [C][C]
  _Float16* kqvh   = (_Float16*)(ws + (size_t)24 * 1024 * 1024);    // 48 MB
  _Float16* atth   = (_Float16*)(ws + (size_t)72 * 1024 * 1024);    // 16 MB

  cvt_f32_to_f16  <<<512, 256, 0, stream>>>(x, xh, Mrows * C);
  cvt_f32_to_f16_T<<<256, 256, 0, stream>>>(w_kqv,  wkqvT,  C, 3 * C);
  cvt_f32_to_f16_T<<<128, 256, 0, stream>>>(w_proj, wprojT, C, C);

  // kqv = x @ w_kqv + b_kqv  (f16 out)
  dim3 g1(3 * C / BN, Mrows / BM);
  gemm_f16_kernel<<<g1, 256, 0, stream>>>(xh, wkqvT, b_kqv, nullptr, kqvh,
                                          Mrows, 3 * C, C, 1);

  // causal flash attention
  attn_kernel<<<Bb * (T / 16), 128, 183488, stream>>>(kqvh, atth, T);

  // out = att @ w_proj + b_proj  (f32 out)
  dim3 g2(C / BN, Mrows / BM);
  gemm_f16_kernel<<<g2, 256, 0, stream>>>(atth, wprojT, b_proj, out, nullptr,
                                          Mrows, C, C, 0);
}
